// E75MultiHead_61529701483346
// MI455X (gfx1250) — matrix operand order repaired and checked by reading, not compiled, not run on hardware
//
#include <hip/hip_runtime.h>
#include <hip/hip_bf16.h>

// ---------------------------------------------------------------------------
// E75 MultiHead delta-rule scan for MI455X (gfx1250, wave32, WMMA).
//
// Phase 1: 5 GEMMs via v_wmma_f32_16x16x32_bf16 with split-bf16 (hi+lo)
//          operands for ~fp32 accuracy. ~103 GFLOP, ~200MB traffic -> ~10us.
// Phase 2: recurrent scan. Rows of S are independent -> 4096 independent
//          wave32 chains (one wave per (b,h,row)), no barriers.
// ---------------------------------------------------------------------------

typedef __attribute__((ext_vector_type(16))) __bf16 v16bf;
typedef __attribute__((ext_vector_type(8)))  __bf16 v8bf;
typedef __attribute__((ext_vector_type(4)))  __bf16 bf4;
typedef __attribute__((ext_vector_type(8)))  float  v8f;
typedef __attribute__((ext_vector_type(4)))  float  f4;

static constexpr int Tt = 2048, Bb = 8, Dd = 1024, Hh = 16, Nn = 32;
static constexpr int BM = 64, BN = 64, BK = 32;
static constexpr int LDT = 40;              // LDS row stride in halves (80B, 16B-aligned chunks)

// Build a 16-half fragment from two contiguous 8-half LDS chunks.
// Elements 0..7 map to VGPR0..3 (first K octet), 8..15 to VGPR4..7 (second).
__device__ inline v16bf frag16(const __bf16* base, int o0, int o1) {
  v8bf lo = *(const v8bf*)(base + o0);
  v8bf hi = *(const v8bf*)(base + o1);
  v16bf r;
#pragma unroll
  for (int e = 0; e < 8; ++e) { r[e] = lo[e]; r[8 + e] = hi[e]; }
  return r;
}

// C[m,n] = sum_k A[m,k] * W[n,k]     A:[M,K] row-major, W:[N,K] row-major.
// Split-bf16: A ~= Ah+Al, W ~= Wh+Wl; C ~= Ah*Wh + Ah*Wl + Al*Wh.
__global__ __launch_bounds__(256)
void e75_gemm_wmma(const float* __restrict__ A, const float* __restrict__ W,
                   float* __restrict__ C, int M, int N, int K) {
  __shared__ __bf16 sAh[BM * LDT], sAl[BM * LDT];
  __shared__ __bf16 sBh[BN * LDT], sBl[BN * LDT];

  const int tid  = threadIdx.x;
  const int lane = tid & 31;
  const int wid  = tid >> 5;         // 8 waves
  const int wr   = wid >> 1;         // wave row 0..3  (16 rows each)
  const int wc   = wid & 1;          // wave col 0..1  (32 cols each)
  const int m0   = blockIdx.y * BM;
  const int n0   = blockIdx.x * BN;

  v8f acc0 = {}; v8f acc1 = {};

  for (int kc = 0; kc < K; kc += BK) {
    // Cooperative load: 64x32 floats for A and for W tiles; float4 per slot.
#pragma unroll
    for (int s = 0; s < 2; ++s) {
      int slot = tid + s * 256;            // 0..511
      int row  = slot >> 3;                // 0..63
      int kq   = slot & 7;                 // float4 index in k
      f4 av = *(const f4*)&A[(size_t)(m0 + row) * K + kc + kq * 4];
      f4 wv = *(const f4*)&W[(size_t)(n0 + row) * K + kc + kq * 4];
      bf4 ah, al, bh, bl;
#pragma unroll
      for (int e = 0; e < 4; ++e) {
        __bf16 h = (__bf16)av[e]; ah[e] = h; al[e] = (__bf16)(av[e] - (float)h);
        __bf16 g = (__bf16)wv[e]; bh[e] = g; bl[e] = (__bf16)(wv[e] - (float)g);
      }
      *(bf4*)&sAh[row * LDT + kq * 4] = ah;
      *(bf4*)&sAl[row * LDT + kq * 4] = al;
      *(bf4*)&sBh[row * LDT + kq * 4] = bh;
      *(bf4*)&sBl[row * LDT + kq * 4] = bl;
    }
    __syncthreads();

    // A fragment: lane<16 holds K {0..7}+{16..23}; lane>=16 holds {8..15}+{24..31}.
    const int mA  = wr * 16 + (lane & 15);
    const int kbA = (lane < 16) ? 0 : 8;
    v16bf ah = frag16(sAh, mA * LDT + kbA, mA * LDT + kbA + 16);
    v16bf al = frag16(sAl, mA * LDT + kbA, mA * LDT + kbA + 16);

    // B fragment: lane<16 holds K {0..7}+{16..23}; lane>=16 holds {8..15}+{24..31}, n = lane%16.
    const int nB0 = wc * 32 + (lane & 15);
    const int nB1 = nB0 + 16;
    const int kbB = (lane < 16) ? 0 : 8;
    v16bf bh0 = frag16(sBh, nB0 * LDT + kbB, nB0 * LDT + kbB + 16);
    v16bf bl0 = frag16(sBl, nB0 * LDT + kbB, nB0 * LDT + kbB + 16);
    v16bf bh1 = frag16(sBh, nB1 * LDT + kbB, nB1 * LDT + kbB + 16);
    v16bf bl1 = frag16(sBl, nB1 * LDT + kbB, nB1 * LDT + kbB + 16);

    acc0 = __builtin_amdgcn_wmma_f32_16x16x32_bf16(false, ah, false, bh0, (short)0, acc0, false, false);
    acc0 = __builtin_amdgcn_wmma_f32_16x16x32_bf16(false, ah, false, bl0, (short)0, acc0, false, false);
    acc0 = __builtin_amdgcn_wmma_f32_16x16x32_bf16(false, al, false, bh0, (short)0, acc0, false, false);
    acc1 = __builtin_amdgcn_wmma_f32_16x16x32_bf16(false, ah, false, bh1, (short)0, acc1, false, false);
    acc1 = __builtin_amdgcn_wmma_f32_16x16x32_bf16(false, ah, false, bl1, (short)0, acc1, false, false);
    acc1 = __builtin_amdgcn_wmma_f32_16x16x32_bf16(false, al, false, bh1, (short)0, acc1, false, false);
    __syncthreads();
  }

  // C/D layout: VGPR r -> M = r (lanes 0-15) or r+8 (lanes 16-31); N = lane%16.
  const int mBase = m0 + wr * 16 + ((lane >> 4) << 3);
  const int nA    = n0 + wc * 32 + (lane & 15);
#pragma unroll
  for (int r = 0; r < 8; ++r) {
    C[(size_t)(mBase + r) * N + nA]      = acc0[r];
    C[(size_t)(mBase + r) * N + nA + 16] = acc1[r];
  }
}

__device__ inline float wave_sum32(float x) {
#pragma unroll
  for (int o = 16; o > 0; o >>= 1) x += __shfl_xor(x, o, 32);
  return x;
}

// One wave per (b,h,row i). Lane j holds S[i][j]. No inter-row dependence.
__global__ __launch_bounds__(256)
void e75_scan(const float* __restrict__ Kp, const float* __restrict__ Vp,
              const float* __restrict__ Qp, const float* __restrict__ Bp,
              const float* __restrict__ S0, const float* __restrict__ bbias,
              float* __restrict__ out, float* __restrict__ Sfin) {
  const int lane = threadIdx.x & 31;
  const int wv   = (blockIdx.x * blockDim.x + threadIdx.x) >> 5;  // 0..B*H*N-1
  const int i    = wv & (Nn - 1);
  const int bh   = wv >> 5;             // 0..B*H-1
  const int h    = bh & (Hh - 1);
  const int b    = bh >> 4;

  float s = S0[((size_t)bh * Nn + i) * Nn + lane];
  const float bb = bbias[h * Nn + i];

  for (int t = 0; t < Tt; ++t) {
    const size_t base = ((size_t)(t * Bb + b) * Hh + h) * Nn;
    float kj = Kp[base + lane];         // coalesced 128B per wave
    float qj = Qp[base + lane];
    float vi = Vp[base + i];            // uniform broadcast
    float bp = Bp[base + i] + bb;
    float beta = 1.f / (1.f + __expf(-bp));

    float n2 = wave_sum32(kj * kj);
    float kn = kj / (sqrtf(n2) + 1e-6f);

    float retr  = wave_sum32(s * kn);
    float delta = vi - retr;

    float xarg = beta * s + delta * kn;
    float xc = fminf(fmaxf(xarg, -15.f), 15.f);   // overflow-safe tanh
    float e2 = __expf(2.f * xc);
    s = (e2 - 1.f) / (e2 + 1.f);

    float sq = wave_sum32(s * qj);
    if (lane == 0) {
      float sig = 1.f / (1.f + __expf(-sq));
      out[base + i] = sq * sq * sig;    // Sq * silu(Sq)
    }
  }
  Sfin[((size_t)bh * Nn + i) * Nn + lane] = s;
}

extern "C" void kernel_launch(void* const* d_in, const int* in_sizes, int n_in,
                              void* d_out, int out_size, void* d_ws, size_t ws_size,
                              hipStream_t stream) {
  const float* x      = (const float*)d_in[0];
  const float* S0     = (const float*)d_in[1];
  const float* W_in   = (const float*)d_in[2];
  const float* W_k    = (const float*)d_in[3];
  const float* W_v    = (const float*)d_in[4];
  const float* W_q    = (const float*)d_in[5];
  const float* W_beta = (const float*)d_in[6];
  const float* b_beta = (const float*)d_in[7];

  const int M  = Tt * Bb;          // 16384
  const int F  = Hh * Nn;          // 512

  float* outp = (float*)d_out;
  float* Sfin = outp + (size_t)Tt * Bb * F;

  // Workspace: xp 64MB + 4 x 32MB = 192MB
  float* xp = (float*)d_ws;
  float* kb = xp + (size_t)M * Dd;
  float* vb = kb + (size_t)M * F;
  float* qb = vb + (size_t)M * F;
  float* pb = qb + (size_t)M * F;

  dim3 blk(256);
  e75_gemm_wmma<<<dim3(Dd / BN, M / BM), blk, 0, stream>>>(x,  W_in,   xp, M, Dd, Dd);
  e75_gemm_wmma<<<dim3(F  / BN, M / BM), blk, 0, stream>>>(xp, W_k,    kb, M, F,  Dd);
  e75_gemm_wmma<<<dim3(F  / BN, M / BM), blk, 0, stream>>>(xp, W_v,    vb, M, F,  Dd);
  e75_gemm_wmma<<<dim3(F  / BN, M / BM), blk, 0, stream>>>(xp, W_q,    qb, M, F,  Dd);
  e75_gemm_wmma<<<dim3(F  / BN, M / BM), blk, 0, stream>>>(xp, W_beta, pb, M, F,  Dd);

  const int nWaves = Bb * Hh * Nn;                 // 4096 independent chains
  e75_scan<<<(nWaves * 32) / 256, blk, 0, stream>>>(kb, vb, qb, pb, S0, b_beta,
                                                    outp, Sfin);
}
